// SimpleGNN_49976239457077
// MI455X (gfx1250) — matrix operand order repaired and checked by reading, not compiled, not run on hardware
//
#include <hip/hip_runtime.h>

typedef __attribute__((ext_vector_type(16))) _Float16 v16h;
typedef __attribute__((ext_vector_type(8)))  float    v8f;

__device__ __forceinline__ void atomAddF(float* p, float v) {
    // no-return agent-scope float add -> global_atomic_add_f32 (no CAS loop)
    __hip_atomic_fetch_add(p, v, __ATOMIC_RELAXED, __HIP_MEMORY_SCOPE_AGENT);
}

// ---------------- degree / dinv ----------------
__global__ void gcn_deg_kernel(const int* __restrict__ dst, float* __restrict__ deg, int E) {
    int t = blockIdx.x * blockDim.x + threadIdx.x;
    if (t < E) atomAddF(&deg[dst[t]], 1.0f);
}

__global__ void gcn_dinv_kernel(float* __restrict__ deg, int N) {
    int t = blockIdx.x * blockDim.x + threadIdx.x;
    if (t < N) deg[t] = rsqrtf(deg[t] + 1.0f);   // self-loop included
}

// ---------------- weight pre-swizzle: f32 row-major -> f16 fragment layout --
// Wh[((ks*NT + t)*32 + lane)*16 + i] = W[(ks*32 + (i<8 ? half*8+i : 16+half*8+i-8))*Nc + t*16 + n]
// (half = lane>>4, n = lane&15) -> a B fragment is one contiguous 32B v16h.
template <int NT>   // Nc = NT*16, NT in {4,8}
__global__ void gcn_wswizzle(const float* __restrict__ W, _Float16* __restrict__ Wh, int K) {
    constexpr int Nc = NT * 16;
    constexpr int LT = (NT == 8) ? 3 : 2;
    int idx = blockIdx.x * blockDim.x + threadIdx.x;
    if (idx >= K * Nc) return;
    int i    = idx & 15;
    int lane = (idx >> 4) & 31;
    int t    = (idx >> 9) & (NT - 1);
    int ks   = idx >> (9 + LT);
    int half = lane >> 4, n = lane & 15;
    Wh[idx] = (_Float16)W[(ks * 32 + ((i < 8) ? (half * 8 + i) : (16 + half * 8 + i - 8))) * Nc + t * 16 + n];
}

// ---------------- GEMM: H[M,Nc] = X[M,K] @ W[K,Nc], f16 WMMA, f32 accum ----
// blockDim = 128 (4 waves). Wave w owns the 16-row strip (blockIdx.x*4+w) and
// computes ALL NT column tiles from one A fragment (X read exactly once).
// B fragments stream from the pre-swizzled f16 weights (L2-resident) with a
// one-fragment software double buffer so each WMMA overlaps the next B load.
template <int NT>   // Nc = NT*16
__global__ void gcn_gemm_wmma(const float* __restrict__ X, const _Float16* __restrict__ Wh,
                              float* __restrict__ H, int M, int K) {
    constexpr int Nc = NT * 16;
    const int wave = threadIdx.x >> 5;
    const int lane = threadIdx.x & 31;
    const int m0   = (blockIdx.x * 4 + wave) * 16;
    if (m0 >= M) return;                    // wave-uniform: EXEC stays all-ones

    const int half = lane >> 4;             // 0: lanes 0-15, 1: lanes 16-31
    const int mn   = lane & 15;             // row (A) / col (B, C)
    const float* rowp = X + (size_t)(m0 + mn) * K;

    v8f acc[NT] = {};

    for (int kk = 0; kk < K; kk += 32) {
        // A 16x32 f16: lane(half,m): K = {half*8+0..7, 16+half*8+0..7}
        v16h a;
#pragma unroll
        for (int i = 0; i < 8; ++i) {
            a[i]     = (_Float16)rowp[kk + half * 8 + i];
            a[i + 8] = (_Float16)rowp[kk + 16 + half * 8 + i];
        }
        const _Float16* wq = Wh + (size_t)(kk >> 5) * (NT * 512) + lane * 16;
        v16h bcur = *(const v16h*)(wq);          // 32B aligned fragment
        v16h bnext;
#pragma unroll
        for (int t = 0; t < NT; ++t) {
            if (t + 1 < NT) bnext = *(const v16h*)(wq + (t + 1) * 512);
            acc[t] = __builtin_amdgcn_wmma_f32_16x16x32_f16(
                false, a, false, bcur, (short)0, acc[t], false, false);
            if (t + 1 < NT) bcur = bnext;
        }
    }

    // C/D: VGPR v holds row (v + 8*half), col mn
#pragma unroll
    for (int t = 0; t < NT; ++t) {
        float* outp = H + (size_t)m0 * Nc + t * 16 + mn;
#pragma unroll
        for (int v = 0; v < 8; ++v)
            outp[(size_t)(v + half * 8) * Nc] = acc[t][v];
    }
}

// ---------------- edge scatter: agg[dst] += h[src] * dinv[src]*dinv[dst] ----
// one wave per edge; float4 streaming + float atomics
__global__ void gcn_scatter_kernel(const float* __restrict__ h, const int* __restrict__ src,
                                   const int* __restrict__ dst, const float* __restrict__ dinv,
                                   float* __restrict__ agg, int E, int D) {
    int gw   = (blockIdx.x * blockDim.x + threadIdx.x) >> 5;
    int lane = threadIdx.x & 31;
    int nw   = (gridDim.x * blockDim.x) >> 5;
    int chunks = D >> 2;
    for (int e = gw; e < E; e += nw) {
        int s = src[e], d = dst[e];
        float nrm = dinv[s] * dinv[d];
        const float4* hp = (const float4*)(h + (size_t)s * D);
        float* ap = agg + (size_t)d * D;
        for (int c = lane; c < chunks; c += 32) {
            float4 v = hp[c];
            atomAddF(ap + 4 * c + 0, v.x * nrm);
            atomAddF(ap + 4 * c + 1, v.y * nrm);
            atomAddF(ap + 4 * c + 2, v.z * nrm);
            atomAddF(ap + 4 * c + 3, v.w * nrm);
        }
    }
}

// ---------------- epilogue: out = act(agg + dinv^2 * h + b) (in place ok) ---
__global__ void gcn_epilogue_kernel(float* __restrict__ agg, const float* __restrict__ h,
                                    const float* __restrict__ dinv, const float* __restrict__ b,
                                    int N, int D, int relu) {
    size_t t = (size_t)blockIdx.x * blockDim.x + threadIdx.x;
    size_t total = (size_t)N * D;
    if (t >= total) return;
    int i = (int)(t / D);
    int j = (int)(t % D);
    float dv  = dinv[i];
    float val = agg[t] + dv * dv * h[t] + b[j];
    if (relu) val = fmaxf(val, 0.0f);
    agg[t] = val;
}

// ---------------- row L2 normalize (D = 64), one wave per node -------------
__global__ void gcn_normalize_kernel(const float* __restrict__ h, float* __restrict__ out, int N) {
    int gw   = (blockIdx.x * blockDim.x + threadIdx.x) >> 5;
    int lane = threadIdx.x & 31;
    if (gw >= N) return;
    const float2* hp = (const float2*)(h + (size_t)gw * 64);
    float2 v = hp[lane];
    float s = v.x * v.x + v.y * v.y;
#pragma unroll
    for (int off = 16; off > 0; off >>= 1) s += __shfl_xor(s, off, 32);
    float denom = fmaxf(sqrtf(s), 1e-12f);
    float2 o; o.x = v.x / denom; o.y = v.y / denom;
    ((float2*)(out + (size_t)gw * 64))[lane] = o;
}

// ---------------------------------------------------------------------------
extern "C" void kernel_launch(void* const* d_in, const int* in_sizes, int n_in,
                              void* d_out, int out_size, void* d_ws, size_t ws_size,
                              hipStream_t stream) {
    const int IN_DIM = 256, HID = 128, EMB = 64;
    const float* x  = (const float*)d_in[0];
    const int*   ei = (const int*)d_in[1];
    const float* W1 = (const float*)d_in[2];
    const float* b1 = (const float*)d_in[3];
    const float* W2 = (const float*)d_in[4];
    const float* b2 = (const float*)d_in[5];
    const float* W3 = (const float*)d_in[6];
    const float* b3 = (const float*)d_in[7];

    const int N = in_sizes[0] / IN_DIM;     // 100000
    const int E = in_sizes[1] / 2;          // 1600000
    const int* src = ei;
    const int* dst = ei + E;

    // workspace layout
    char* ws = (char*)d_ws;
    size_t bufBytes = (size_t)N * HID * sizeof(float);          // 51.2 MB
    float* dinv = (float*)ws;
    float* B0 = (float*)(ws + ((size_t)N * sizeof(float) + 511) / 512 * 512);
    float* B1 = (float*)((char*)B0 + bufBytes);
    float* B2 = (float*)((char*)B1 + bufBytes);
    _Float16* Wh1 = (_Float16*)((char*)B2 + bufBytes);          // 64 KB
    _Float16* Wh2 = Wh1 + (size_t)IN_DIM * HID;                 // 32 KB
    _Float16* Wh3 = Wh2 + (size_t)HID * HID;                    // 16 KB

    // --- weight pre-swizzle (tiny, once per launch) ---
    gcn_wswizzle<8><<<(IN_DIM * HID + 255) / 256, 256, 0, stream>>>(W1, Wh1, IN_DIM);
    gcn_wswizzle<8><<<(HID * HID + 255) / 256, 256, 0, stream>>>(W2, Wh2, HID);
    gcn_wswizzle<4><<<(HID * EMB + 255) / 256, 256, 0, stream>>>(W3, Wh3, HID);

    // --- degrees -> dinv ---
    hipMemsetAsync(dinv, 0, (size_t)N * sizeof(float), stream);
    gcn_deg_kernel<<<(E + 255) / 256, 256, 0, stream>>>(dst, dinv, E);
    gcn_dinv_kernel<<<(N + 255) / 256, 256, 0, stream>>>(dinv, N);

    const int tilesM   = (N + 15) / 16;                          // 6250
    const int gemmGx   = (tilesM + 3) / 4;
    const int scatBlks = (E + 7) / 8;                            // 8 waves/block

    // ---------------- layer 1: 256 -> 128, ReLU ----------------
    gcn_gemm_wmma<8><<<gemmGx, 128, 0, stream>>>(x, Wh1, B0, N, IN_DIM);
    hipMemsetAsync(B1, 0, bufBytes, stream);
    gcn_scatter_kernel<<<scatBlks, 256, 0, stream>>>(B0, src, dst, dinv, B1, E, HID);
    gcn_epilogue_kernel<<<(int)(((size_t)N * HID + 255) / 256), 256, 0, stream>>>(
        B1, B0, dinv, b1, N, HID, 1);

    // ---------------- layer 2: 128 -> 128, ReLU ----------------
    gcn_gemm_wmma<8><<<gemmGx, 128, 0, stream>>>(B1, Wh2, B0, N, HID);
    hipMemsetAsync(B2, 0, bufBytes, stream);
    gcn_scatter_kernel<<<scatBlks, 256, 0, stream>>>(B0, src, dst, dinv, B2, E, HID);
    gcn_epilogue_kernel<<<(int)(((size_t)N * HID + 255) / 256), 256, 0, stream>>>(
        B2, B0, dinv, b2, N, HID, 1);

    // ---------------- layer 3: 128 -> 64, no activation --------
    gcn_gemm_wmma<4><<<gemmGx, 128, 0, stream>>>(B2, Wh3, B0, N, HID);
    hipMemsetAsync(B1, 0, (size_t)N * EMB * sizeof(float), stream);
    gcn_scatter_kernel<<<scatBlks, 256, 0, stream>>>(B0, src, dst, dinv, B1, E, EMB);
    gcn_epilogue_kernel<<<(int)(((size_t)N * EMB + 255) / 256), 256, 0, stream>>>(
        B1, B0, dinv, b3, N, EMB, 0);

    // ---------------- row-wise L2 normalize --------------------
    gcn_normalize_kernel<<<(N + 7) / 8, 256, 0, stream>>>(B1, (float*)d_out, N);
}
